// LSTM1_51591147160292
// MI455X (gfx1250) — matrix operand order repaired and checked by reading, not compiled, not run on hardware
//
#include <hip/hip_runtime.h>
#include <hip/hip_bf16.h>
#include <stdint.h>

typedef __attribute__((ext_vector_type(16))) _Float16 v16h;
typedef __attribute__((ext_vector_type(8)))  _Float16 v8h;
typedef __attribute__((ext_vector_type(8)))  float    v8f;
typedef __attribute__((ext_vector_type(4)))  uint32_t v4u;
typedef __attribute__((ext_vector_type(8)))  int      v8i;
typedef __attribute__((ext_vector_type(4)))  int      v4i;

#define LSTM_B 256
#define LSTM_T 4096
#define LSTM_I 6
#define LSTM_H 64
#define LSTM_L 4
// combined K = [h_in(64) | h_rec(64)] = 128 ; gates along N = 4H = 256

#if defined(__has_builtin)
#if __has_builtin(__builtin_amdgcn_tensor_load_to_lds) && __has_builtin(__builtin_amdgcn_s_wait_tensorcnt)
#define HAVE_TDM 1
#endif
#endif
#ifndef HAVE_TDM
#define HAVE_TDM 0
#endif

// Probe: fire a diagnostic ONLY if the gfx1250 device pass lacks the TDM builtin
// (host pass never defines __gfx1250__, so a clean stderr == TDM is in the device code).
#if defined(__gfx1250__) && !HAVE_TDM
#warning "gfx1250 device pass: tensor_load_to_lds builtin NOT available - fallback copy path in use"
#endif

__device__ __forceinline__ float sigf(float x) {
  return 1.0f / (1.0f + __expf(-x));
}
__device__ __forceinline__ float tanh_f(float x) {
  float e = __expf(-2.0f * fabsf(x));
  float r = (1.0f - e) / (1.0f + e);
  return x >= 0.0f ? r : -r;
}

// A-fragment (16x32 f16): lane group g holds K {k0+g*8..+7} then {k0+16+g*8..+7}
__device__ __forceinline__ v16h frag_interleave(const _Float16* p1, const _Float16* p2) {
  v8h a = *(const v8h*)p1;
  v8h b = *(const v8h*)p2;
  return __builtin_shufflevector(a, b, 0,1,2,3,4,5,6,7,8,9,10,11,12,13,14,15);
}

#if HAVE_TDM
// 1-D TDM copy: 16384 f16 elements (32 KB) global -> LDS
__device__ __forceinline__ void tdm_load_32k(uint32_t lds_off, const void* gptr) {
  uint64_t ga = (uint64_t)(uintptr_t)gptr;
  v4u g0;
  g0[0] = 1u;                                            // count=1, user mode
  g0[1] = lds_off;                                       // LDS dest byte address
  g0[2] = (uint32_t)ga;                                  // global_addr[31:0]
  g0[3] = ((uint32_t)(ga >> 32) & 0x01FFFFFFu) | (2u << 30); // addr[56:32] | type=2
  v8i g1;
  g1[0] = 0x00010000;        // data_size=1 (2 bytes), no mask/pad/iterate
  g1[1] = (int)0x40000000;   // tensor_dim0 = 16384 (low 16 bits at [31:16])
  g1[2] = 0x00010000;        // tensor_dim0 hi = 0 ; tensor_dim1 = 1
  g1[3] = (int)0x40000000;   // tile_dim0 = 16384
  g1[4] = 0;                 // tile_dim1 = 0 (unused), tile_dim2 = 0
  g1[5] = 16384;             // tensor_dim0_stride lo
  g1[6] = (int)0x40000000;   // stride0 hi = 0 ; tensor_dim1_stride lo = 16384
  g1[7] = 0;
  v4i z4 = {0, 0, 0, 0};
#if __clang_major__ >= 23
  v8i z8 = {0, 0, 0, 0, 0, 0, 0, 0};
  __builtin_amdgcn_tensor_load_to_lds(g0, g1, z4, z4, z8, 0);
#else
  __builtin_amdgcn_tensor_load_to_lds(g0, g1, z4, z4, 0);
#endif
}
#endif

// Persistent one-WGP LSTM layer: fused g = [h_in|h_prev] @ W^T + bias, gates, c/h update.
// hin_g/hout_g: [T][256][64] f16 ; Wg: [256 gates][128 K] f16 ; bias: [256] f32
__global__ void __launch_bounds__(1024, 1)
lstm_layer_kernel(const _Float16* __restrict__ hin_g,
                  _Float16* __restrict__ hout_g,
                  const _Float16* __restrict__ Wg,
                  const float* __restrict__ bias) {
  extern __shared__ _Float16 smem[];
  _Float16* hbuf0 = smem;                 // 256*64 f16 (prev-layer h, buffer 0)
  _Float16* hbuf1 = smem + 256 * 64;      // buffer 1
  _Float16* hrec  = smem + 2 * 256 * 64;  // this layer's h_{t-1}
  _Float16* Wlds  = smem + 3 * 256 * 64;  // 256*128 f16 weights

  const int tid  = threadIdx.x;
  const int lane = tid & 31;
  const int w    = tid >> 5;    // wave 0..31
  const int mt   = w & 15;      // M tile (batch rows mt*16..+15)
  const int nh   = w >> 4;      // half of H columns (0..1)
  const int lrow = lane & 15;
  const int lgrp = lane >> 4;

  for (int i = tid; i < 256 * 128; i += 1024) Wlds[i] = Wg[i];
  for (int i = tid; i < 256 * 64; i += 1024) hrec[i] = (_Float16)0.0f;

  // per-lane gate biases (scalars only; added post-WMMA to keep VGPR pressure low)
  float bv[8];
#pragma unroll
  for (int g = 0; g < 4; ++g)
#pragma unroll
    for (int j = 0; j < 2; ++j)
      bv[g * 2 + j] = bias[g * 64 + nh * 32 + j * 16 + lrow];

  v8f cst[2] = {};   // cell state for this wave's 16x32 patch (f32, persistent)

#if HAVE_TDM
  const uint32_t lds_off0 = (uint32_t)(uintptr_t)(void*)hbuf0;
  const uint32_t lds_off1 = (uint32_t)(uintptr_t)(void*)hbuf1;
  if (w == 0) {
    tdm_load_32k(lds_off0, hin_g);                 // t = 0
    __builtin_amdgcn_s_wait_tensorcnt(0);
  }
#else
  for (int i = tid; i < 256 * 64; i += 1024) hbuf0[i] = hin_g[i];
#endif
  __syncthreads();

  const int arow = mt * 16 + lrow;

#pragma unroll 1
  for (int t = 0; t < LSTM_T; ++t) {
    const _Float16* hin = (t & 1) ? hbuf1 : hbuf0;
#if HAVE_TDM
    if (w == 0 && t + 1 < LSTM_T)                  // prefetch next timestep
      tdm_load_32k((t & 1) ? lds_off0 : lds_off1,
                   hin_g + (size_t)(t + 1) * 256 * 64);
#endif
    // ---- A fragments for this wave's M tile (shared by all 8 N tiles) ----
    v16h af[4];
    {
      const _Float16* r0 = hin + arow * 64;
      const _Float16* r1 = hrec + arow * 64;
      af[0] = frag_interleave(r0 + lgrp * 8,      r0 + 16 + lgrp * 8);
      af[1] = frag_interleave(r0 + 32 + lgrp * 8, r0 + 48 + lgrp * 8);
      af[2] = frag_interleave(r1 + lgrp * 8,      r1 + 16 + lgrp * 8);
      af[3] = frag_interleave(r1 + 32 + lgrp * 8, r1 + 48 + lgrp * 8);
    }

    v8f hv[2];
#pragma unroll
    for (int j = 0; j < 2; ++j) {
      v8f acc[4];
#pragma unroll
      for (int g = 0; g < 4; ++g) {
        const int n = (g * 4 + nh * 2 + j) * 16 + lrow;    // gate column
        const _Float16* wr = Wlds + n * 128 + lgrp * 8;    // B frag: K {g*8..+7} then {16+g*8..+7}
        v8f a = {};                                        // zero C; bias added later
#pragma unroll
        for (int ks = 0; ks < 4; ++ks) {
          v16h bf = frag_interleave(wr + ks * 32, wr + ks * 32 + 16);
          a = __builtin_amdgcn_wmma_f32_16x16x32_f16(
                  false, af[ks], false, bf, (short)0, a, false, false);
        }
        acc[g] = a;
      }
      // gate order i, f, g~, o (PyTorch); bias folded in here (scalar per lane)
      const float bi = bv[0 * 2 + j], bf_ = bv[1 * 2 + j];
      const float bg = bv[2 * 2 + j], bo  = bv[3 * 2 + j];
#pragma unroll
      for (int e = 0; e < 8; ++e) {
        float ig = sigf(acc[0][e] + bi);
        float fg = sigf(acc[1][e] + bf_);
        float gg = tanh_f(acc[2][e] + bg);
        float og = sigf(acc[3][e] + bo);
        float cc = fg * cst[j][e] + ig * gg;
        cst[j][e] = cc;
        hv[j][e] = og * tanh_f(cc);
      }
    }

    // stream h to global for the next layer (no one reads it this launch)
    {
      _Float16* og = hout_g + (size_t)t * 256 * 64;
#pragma unroll
      for (int j = 0; j < 2; ++j) {
        const int ncol = nh * 32 + j * 16 + lrow;
#pragma unroll
        for (int e = 0; e < 8; ++e) {
          const int brow = mt * 16 + lgrp * 8 + e;
          og[brow * 64 + ncol] = (_Float16)hv[j][e];
        }
      }
    }

    __syncthreads();   // all WMMA reads of hrec complete before overwrite

#pragma unroll
    for (int j = 0; j < 2; ++j) {
      const int ncol = nh * 32 + j * 16 + lrow;
#pragma unroll
      for (int e = 0; e < 8; ++e) {
        const int brow = mt * 16 + lgrp * 8 + e;
        hrec[brow * 64 + ncol] = (_Float16)hv[j][e];
      }
    }

#if HAVE_TDM
    if (w == 0) __builtin_amdgcn_s_wait_tensorcnt(0);   // prefetch landed
#else
    if (t + 1 < LSTM_T) {
      const _Float16* src = hin_g + (size_t)(t + 1) * 256 * 64;
      _Float16* dst = (t & 1) ? hbuf0 : hbuf1;
      for (int i = tid; i < 256 * 64; i += 1024) dst[i] = src[i];
    }
#endif
    __syncthreads();   // hrec + next hin visible to everyone
  }
}

// layer-0 input: pad x [B,T,6] (batch-first) into time-major f16 [T][B][64]
__global__ void prep_x_kernel(const float* __restrict__ x, _Float16* __restrict__ seq) {
  int idx = blockIdx.x * blockDim.x + threadIdx.x;   // t*B + b
  if (idx >= LSTM_T * LSTM_B) return;
  int t = idx / LSTM_B;
  int b = idx % LSTM_B;
  _Float16* dst = seq + (size_t)idx * 64;
  const float* src = x + ((size_t)b * LSTM_T + t) * LSTM_I;
#pragma unroll
  for (int k = 0; k < 64; ++k)
    dst[k] = (k < LSTM_I) ? (_Float16)src[k] : (_Float16)0.0f;
}

// build per-layer fused weights Wc[l][256][128] = [Wih(pad to 64) | Whh] in f16, bias=bih+bhh
__global__ void prep_w_kernel(const float* __restrict__ Wih0, const float* __restrict__ Wih_rest,
                              const float* __restrict__ Whh, const float* __restrict__ bih,
                              const float* __restrict__ bhh,
                              _Float16* __restrict__ Wc, float* __restrict__ bias) {
  int idx = blockIdx.x * blockDim.x + threadIdx.x;
  if (idx >= 4 * 256 * 128) return;
  int k = idx & 127;
  int g = (idx >> 7) & 255;
  int l = idx >> 15;
  float v;
  if (k < 64) {
    if (l == 0) v = (k < LSTM_I) ? Wih0[g * LSTM_I + k] : 0.0f;
    else        v = Wih_rest[(((l - 1) * 256) + g) * 64 + k];
  } else {
    v = Whh[((l * 256) + g) * 64 + (k - 64)];
  }
  Wc[idx] = (_Float16)v;
  if (k == 0) bias[l * 256 + g] = bih[l * 256 + g] + bhh[l * 256 + g];
}

// out[b] = fc_b + sum_k relu(h_last[b][k]) * fc_w[k]
__global__ void fc_kernel(const _Float16* __restrict__ hseq, const float* __restrict__ fc_w,
                          const float* __restrict__ fc_b, float* __restrict__ out) {
  int b = threadIdx.x;
  if (b >= LSTM_B) return;
  const _Float16* h = hseq + ((size_t)(LSTM_T - 1) * 256 + b) * 64;
  float s = fc_b[0];
#pragma unroll
  for (int k = 0; k < 64; ++k) {
    float v = (float)h[k];
    v = v > 0.0f ? v : 0.0f;
    s += v * fc_w[k];
  }
  out[b] = s;
}

extern "C" void kernel_launch(void* const* d_in, const int* in_sizes, int n_in,
                              void* d_out, int out_size, void* d_ws, size_t ws_size,
                              hipStream_t stream) {
  (void)in_sizes; (void)n_in; (void)out_size; (void)ws_size;
  const float* x        = (const float*)d_in[0];
  const float* Wih0     = (const float*)d_in[1];
  const float* Wih_rest = (const float*)d_in[2];
  const float* Whh      = (const float*)d_in[3];
  const float* bih      = (const float*)d_in[4];
  const float* bhh      = (const float*)d_in[5];
  const float* fc_w     = (const float*)d_in[6];
  const float* fc_b     = (const float*)d_in[7];
  float* out = (float*)d_out;

  // workspace: two ping-pong f16 sequences (134 MB each) + fused weights + bias
  const size_t seq_elems = (size_t)LSTM_T * 256 * 64;
  _Float16* seqX = (_Float16*)d_ws;
  _Float16* seqY = seqX + seq_elems;
  _Float16* Wc   = seqY + seq_elems;
  float*    bias = (float*)(Wc + 4 * 256 * 128);

  prep_x_kernel<<<dim3((LSTM_T * LSTM_B + 255) / 256), dim3(256), 0, stream>>>(x, seqX);
  prep_w_kernel<<<dim3((4 * 256 * 128 + 255) / 256), dim3(256), 0, stream>>>(
      Wih0, Wih_rest, Whh, bih, bhh, Wc, bias);

  const size_t smem_bytes = (size_t)(3 * 256 * 64 + 256 * 128) * sizeof(_Float16); // 160 KB
  for (int l = 0; l < LSTM_L; ++l) {
    const _Float16* li = (l & 1) ? seqY : seqX;
    _Float16*       lo = (l & 1) ? seqX : seqY;
    lstm_layer_kernel<<<dim3(1), dim3(1024), smem_bytes, stream>>>(
        li, lo, Wc + (size_t)l * 256 * 128, bias + (size_t)l * 256);
  }
  // L0: X->Y, L1: Y->X, L2: X->Y, L3: Y->X  => final hidden sequence in seqX
  fc_kernel<<<dim3(1), dim3(256), 0, stream>>>(seqX, fc_w, fc_b, out);
}
